// MultiScaleRetention_57131654971864
// MI455X (gfx1250) — matrix operand order repaired and checked by reading, not compiled, not run on hardware
//
#include <hip/hip_runtime.h>
#include <hip/hip_bf16.h>

// ---------------------------------------------------------------------------
// MultiScaleRetention forward for MI455X (gfx1250), wave32 + WMMA bf16 + TDM.
// Double-buffered LDS tiles: TDM load for step i+1 overlaps WMMA of step i.
// B=4, S=2048, C=512, H=8, F=2, KD=64, HD=128, CF=1024.
// ---------------------------------------------------------------------------

#define S_LEN 2048
#define C_DIM 512
#define H_NUM 8
#define KD_DIM 64
#define HD_DIM 128
#define CF_DIM 1024
#define B_NUM 4
#define M_ROWS (B_NUM * S_LEN)   // 8192

typedef __bf16 bf16_t;
typedef __attribute__((ext_vector_type(16))) __bf16 v16bf;
typedef __attribute__((ext_vector_type(8)))  __bf16 v8bf;
typedef __attribute__((ext_vector_type(8)))  float  v8f;
typedef __attribute__((ext_vector_type(4)))  unsigned int u32x4;
typedef __attribute__((ext_vector_type(4)))  int i32x4;
typedef __attribute__((ext_vector_type(8)))  int i32x8;

__device__ __forceinline__ float gelu_f(float x) {
    return 0.5f * x * (1.0f + erff(x * 0.70710678118654752f));
}

// Build a 16x32 bf16 A-fragment for one lane from two 16-byte chunks.
// Contiguous layout (matches B): lanes 0-15 hold K {0..15}; lanes 16-31 hold K {16..31}.
__device__ __forceinline__ v16bf make_afrag(const bf16_t* p0, const bf16_t* p1) {
    v8bf lo = *(const v8bf*)p0;
    v8bf hi = *(const v8bf*)p1;
    v16bf a;
#pragma unroll
    for (int i = 0; i < 8; ++i) { a[i] = lo[i]; a[8 + i] = hi[i]; }
    return a;
}

#define WMMA_BF16(A, Bm, Cm) \
    __builtin_amdgcn_wmma_f32_16x16x32_bf16(false, (A), false, (Bm), (short)0, (Cm), false, false)

// ---------------------------------------------------------------------------
// Tensor Data Mover helpers (D# per CDNA5 ISA ch.8).
// Group0: [1:0]=count(1) [63:32]=lds_addr [120:64]=global_addr [127:126]=type(2)
// Group1: [17:16]=data_size(1 -> 2 bytes) [79:48]=tensor_dim0 [111:80]=tensor_dim1
//         [127:112]=tile_dim0 [143:128]=tile_dim1 [207:160]=tensor_dim0_stride
// Groups 2/3: zero (2D tile; tile_dim2..4 = 0 -> unused).
// ---------------------------------------------------------------------------
__device__ __forceinline__ u32x4 tdm_g0(unsigned int lds_off, const void* gaddr) {
    unsigned long long ga = (unsigned long long)(uintptr_t)gaddr;
    u32x4 g;
    g[0] = 1u;                                                   // count = 1
    g[1] = lds_off;                                              // lds_addr
    g[2] = (unsigned int)ga;                                     // global_addr[31:0]
    g[3] = ((unsigned int)(ga >> 32) & 0x01FFFFFFu) | 0x80000000u; // addr[56:32] | type=2
    return g;
}

__device__ __forceinline__ i32x8 tdm_g1_2d(unsigned int dim0, unsigned int dim1,
                                           unsigned int tile0, unsigned int tile1,
                                           unsigned long long stride0) {
    i32x8 g;
    g[0] = (int)(1u << 16);                                      // data_size = 2 bytes
    g[1] = (int)((dim0 & 0xFFFFu) << 16);                        // dim0 low16 at [63:48]
    g[2] = (int)((dim0 >> 16) | ((dim1 & 0xFFFFu) << 16));       // dim0 hi16 | dim1 low16
    g[3] = (int)((dim1 >> 16) | (tile0 << 16));                  // dim1 hi16 | tile_dim0
    g[4] = (int)tile1;                                           // tile_dim1 | tile_dim2=0
    g[5] = (int)(unsigned int)stride0;                           // stride0[31:0]
    g[6] = (int)((unsigned int)(stride0 >> 32) & 0xFFFFu);       // stride0[47:32] | stride1 lo=0
    g[7] = 0;                                                    // stride1 hi = 0
    return g;
}

__device__ __forceinline__ void tdm_load_2d(u32x4 g0, i32x8 g1) {
    i32x4 z4 = {0, 0, 0, 0};
#if defined(__clang_major__) && (__clang_major__ >= 23)
    i32x8 z8 = {0, 0, 0, 0, 0, 0, 0, 0};
    __builtin_amdgcn_tensor_load_to_lds(g0, g1, z4, z4, z8, 0);
#else
    __builtin_amdgcn_tensor_load_to_lds(g0, g1, z4, z4, 0);
#endif
}

__device__ __forceinline__ unsigned int lds_offset_of(const void* p) {
    return (unsigned int)(uintptr_t)p;  // LDS aperture: offset = addr[31:0]
}

// ---------------------------------------------------------------------------
// Conversion kernels
// ---------------------------------------------------------------------------
__global__ void cvt_f32_bf16(const float* __restrict__ in, bf16_t* __restrict__ out, int n) {
    int i = blockIdx.x * 256 + threadIdx.x;
    if (i < n) out[i] = (bf16_t)in[i];
}

// WT[n*K + k] = W[k*N + n]  (store weight transposed in bf16)
__global__ void cvt_transpose_bf16(const float* __restrict__ W, bf16_t* __restrict__ WT,
                                   int K, int N) {
    int i = blockIdx.x * 256 + threadIdx.x;
    if (i < K * N) {
        int nn = i / K, kk = i % K;
        WT[i] = (bf16_t)W[(size_t)kk * N + nn];
    }
}

// ---------------------------------------------------------------------------
// Generic WMMA GEMM: acc = A(MxK,bf16,row-major) * BT(NxK,bf16,row-major)^T
// Double-buffered 128x64 A / 64x64 BT tiles fetched by the Tensor Data Mover:
// TDM for step i+1 is issued right after the barrier publishing step i, so it
// overlaps the 8 WMMAs of step i. One barrier per k-step.
// Epilogue modes:
//  0: q-proj  -> gelu+bias, rotary, store bf16 qr[b,h,s,kd]
//  1: k-proj  -> gelu+bias, *KD^-0.5, rotary, store bf16 kr[b,h,s,kd]
//  2: v-proj  -> gelu+bias, store bf16 vrT[b,h,d,s]
//  3: g-proj  -> gelu+bias, store f32 g[m,n]
//  4: o-proj  -> gelu+bias, store f32 d_out[m,n]
// ---------------------------------------------------------------------------
__global__ __launch_bounds__(256)
void gemm_bf16_wmma(const bf16_t* __restrict__ A, const bf16_t* __restrict__ BT,
                    const float* __restrict__ bias, int M, int N, int K, int mode,
                    const float* __restrict__ sinp, const float* __restrict__ cosp,
                    bf16_t* __restrict__ outb, float* __restrict__ outf) {
    __shared__ __attribute__((aligned(64))) bf16_t As[2][128 * 64];
    __shared__ __attribute__((aligned(64))) bf16_t Bs[2][64 * 64];

    const int tid  = threadIdx.x;
    const int lane = tid & 31;
    const int wave = tid >> 5;
    const int half = lane >> 4;
    const int l    = lane & 15;
    const int wm   = wave >> 1;      // 0..3  (M sub-block of 32)
    const int wn   = wave & 1;       // 0..1  (N sub-block of 32)
    const int m0   = blockIdx.y * 128;
    const int n0   = blockIdx.x * 64;
    const int aOff = half ? 16 : 0;  // A-fragment K chunk offsets {aOff, aOff+8}
    const int bOff = half ? 16 : 0;  // B-fragment contiguous K offset

    const i32x8 g1A = tdm_g1_2d((unsigned)K, (unsigned)M, 64u, 128u, (unsigned long long)K);
    const i32x8 g1B = tdm_g1_2d((unsigned)K, (unsigned)N, 64u, 64u,  (unsigned long long)K);
    const bf16_t* aTile = A + (size_t)m0 * K;
    const bf16_t* bTile = BT + (size_t)n0 * K;

    const int nIter = K >> 6;
    v8f acc[2][2] = {};

    if (wave == 0) {
        tdm_load_2d(tdm_g0(lds_offset_of(As[0]), aTile), g1A);
        tdm_load_2d(tdm_g0(lds_offset_of(Bs[0]), bTile), g1B);
    }

    for (int it = 0; it < nIter; ++it) {
        if (wave == 0) __builtin_amdgcn_s_wait_tensorcnt(0);
        __syncthreads();  // publish buf[it&1]; all waves done with buf[(it+1)&1]
        if (wave == 0 && it + 1 < nIter) {
            int k1 = (it + 1) << 6;
            tdm_load_2d(tdm_g0(lds_offset_of(As[(it + 1) & 1]), aTile + k1), g1A);
            tdm_load_2d(tdm_g0(lds_offset_of(Bs[(it + 1) & 1]), bTile + k1), g1B);
        }
        const bf16_t* Ab = As[it & 1];
        const bf16_t* Bb = Bs[it & 1];
#pragma unroll
        for (int ks = 0; ks < 2; ++ks) {
            v16bf af[2], bfm[2];
#pragma unroll
            for (int i = 0; i < 2; ++i) {
                int row = wm * 32 + i * 16 + l;
                af[i] = make_afrag(&Ab[row * 64 + ks * 32 + aOff],
                                   &Ab[row * 64 + ks * 32 + aOff + 8]);
            }
#pragma unroll
            for (int j = 0; j < 2; ++j) {
                int row = wn * 32 + j * 16 + l;
                bfm[j] = *(const v16bf*)&Bb[row * 64 + ks * 32 + bOff];
            }
#pragma unroll
            for (int i = 0; i < 2; ++i)
#pragma unroll
                for (int j = 0; j < 2; ++j)
                    acc[i][j] = WMMA_BF16(af[i], bfm[j], acc[i][j]);
        }
    }

    // Epilogue. C-fragment: col n = base + (lane&15); rows r (+8 for upper half).
#pragma unroll
    for (int i = 0; i < 2; ++i) {
#pragma unroll
        for (int j = 0; j < 2; ++j) {
            v8f c = acc[i][j];
            int mBase = m0 + wm * 32 + i * 16 + half * 8;
            int n = n0 + wn * 32 + j * 16 + l;
            float bv = bias[n];
#pragma unroll
            for (int r = 0; r < 8; ++r) {
                int m = mBase + r;
                float val = gelu_f(c[r] + bv);
                if (mode <= 1) {
                    if (mode == 1) val *= 0.125f;  // KD^-0.5
                    // rotary: pairs (2i,2i+1) live in adjacent lanes (col parity == lane parity)
                    float part = __shfl_xor(val, 1, 32);
                    float rot = ((n & 1) == 0) ? -part : part;
                    int s = m & (S_LEN - 1);
                    int b = m >> 11;
                    int kdi = n & (KD_DIM - 1);
                    int h = n >> 6;
                    float res = val * cosp[s * KD_DIM + kdi] + rot * sinp[s * KD_DIM + kdi];
                    outb[(((size_t)(b * H_NUM + h) * S_LEN + s) << 6) + kdi] = (bf16_t)res;
                } else if (mode == 2) {
                    int s = m & (S_LEN - 1);
                    int b = m >> 11;
                    int d = n & (HD_DIM - 1);
                    int h = n >> 7;
                    outb[((size_t)(b * H_NUM + h) * HD_DIM + d) * S_LEN + s] = (bf16_t)val;
                } else {
                    outf[(size_t)m * N + n] = val;
                }
            }
        }
    }
}

// ---------------------------------------------------------------------------
// Retention attention: per (b,h), 64 rows per block (4 waves x 16 rows).
// Pass 1: row sums of qk*decay -> den.  Pass 2: out = (qk*decay/den) @ v.
// Decay mask computed analytically: gamma_h^(s-t) for s>=t else 0.
// kr / vrT tiles double-buffered via the Tensor Data Mover.
// ---------------------------------------------------------------------------
__global__ __launch_bounds__(128)
void attn_kernel(const bf16_t* __restrict__ qr, const bf16_t* __restrict__ kr,
                 const bf16_t* __restrict__ vrT, float* __restrict__ outw) {
    __shared__ __attribute__((aligned(64))) bf16_t krs[2][64 * 64];    // 64 t x 64 kd
    __shared__ __attribute__((aligned(64))) bf16_t vrs[2][128 * 64];   // 128 d x 64 t
    __shared__ __attribute__((aligned(64))) bf16_t wst[4][16 * 32];    // per-wave w staging

    const int tid  = threadIdx.x;
    const int lane = tid & 31;
    const int wave = tid >> 5;
    const int half = lane >> 4;
    const int l    = lane & 15;
    const int bh   = blockIdx.y;       // b*H + h
    const int h    = bh & (H_NUM - 1);
    const int s0   = blockIdx.x * 64 + wave * 16;
    const int sMaxWave = s0 + 15;
    const int nT   = blockIdx.x + 1;   // number of 64-wide causal t-tiles
    const int aOff = half ? 16 : 0;
    const int bOff = half ? 16 : 0;

    const float gamma = 1.0f - exp2f(-5.0f - (float)h);
    const float lg = log2f(gamma);

    // q A-fragments (16 rows x 64 kd = two 16x32 fragments), held in registers
    const bf16_t* qbase = qr + ((size_t)bh * S_LEN + s0) * KD_DIM;
    v16bf qf0 = make_afrag(qbase + l * 64 + aOff,      qbase + l * 64 + aOff + 8);
    v16bf qf1 = make_afrag(qbase + l * 64 + 32 + aOff, qbase + l * 64 + 32 + aOff + 8);

    const bf16_t* kbase = kr  + (size_t)bh * S_LEN * KD_DIM;
    const bf16_t* vbase = vrT + (size_t)bh * HD_DIM * S_LEN;

    const i32x8 g1K = tdm_g1_2d(64u, (unsigned)S_LEN, 64u, 64u, 64ull);
    const i32x8 g1V = tdm_g1_2d((unsigned)S_LEN, 128u, 64u, 128u, (unsigned long long)S_LEN);

    float rowAcc[8];
#pragma unroll
    for (int r = 0; r < 8; ++r) rowAcc[r] = 0.0f;

    // ---------------- Pass 1: denominators ----------------
    if (wave == 0) tdm_load_2d(tdm_g0(lds_offset_of(krs[0]), kbase), g1K);
    for (int it = 0; it < nT; ++it) {
        if (wave == 0) __builtin_amdgcn_s_wait_tensorcnt(0);
        __syncthreads();
        if (wave == 0 && it + 1 < nT)
            tdm_load_2d(tdm_g0(lds_offset_of(krs[(it + 1) & 1]),
                               kbase + ((size_t)(it + 1) << 6) * 64), g1K);
        const bf16_t* kb = krs[it & 1];
        int t0 = it << 6;
#pragma unroll
        for (int tt = 0; tt < 4; ++tt) {
            int tb = t0 + tt * 16;
            if (tb <= sMaxWave) {  // wave-uniform causal skip
                v16bf kf0 = *(const v16bf*)&kb[(tt * 16 + l) * 64 + bOff];
                v16bf kf1 = *(const v16bf*)&kb[(tt * 16 + l) * 64 + 32 + bOff];
                v8f c = {};
                c = WMMA_BF16(qf0, kf0, c);
                c = WMMA_BF16(qf1, kf1, c);
                int t = tb + l;
#pragma unroll
                for (int r = 0; r < 8; ++r) {
                    int s = s0 + half * 8 + r;
                    if (s >= t) rowAcc[r] += c[r] * exp2f((float)(s - t) * lg);
                }
            }
        }
    }
    // reduce across the 16 lanes that share each row
#pragma unroll
    for (int msk = 1; msk < 16; msk <<= 1)
#pragma unroll
        for (int r = 0; r < 8; ++r) rowAcc[r] += __shfl_xor(rowAcc[r], msk, 32);

    float invDen[8];
#pragma unroll
    for (int r = 0; r < 8; ++r) {
        float a = fabsf(rowAcc[r]);
        a = (a < 1.0f) ? 1.0f : a;
        invDen[r] = 1.0f / (a + 1e-6f);
    }

    __syncthreads();  // pass-1 reads finished on all waves before pass-2 TDM overwrites

    // ---------------- Pass 2: output ----------------
    v8f accD[8] = {};
    if (wave == 0) {
        tdm_load_2d(tdm_g0(lds_offset_of(krs[0]), kbase), g1K);
        tdm_load_2d(tdm_g0(lds_offset_of(vrs[0]), vbase), g1V);
    }
    for (int it = 0; it < nT; ++it) {
        if (wave == 0) __builtin_amdgcn_s_wait_tensorcnt(0);
        __syncthreads();
        if (wave == 0 && it + 1 < nT) {
            tdm_load_2d(tdm_g0(lds_offset_of(krs[(it + 1) & 1]),
                               kbase + ((size_t)(it + 1) << 6) * 64), g1K);
            tdm_load_2d(tdm_g0(lds_offset_of(vrs[(it + 1) & 1]),
                               vbase + ((size_t)(it + 1) << 6)), g1V);
        }
        const bf16_t* kb = krs[it & 1];
        const bf16_t* vb = vrs[it & 1];
        int t0 = it << 6;
#pragma unroll
        for (int pair = 0; pair < 2; ++pair) {
            int tb0 = t0 + pair * 32;
            if (tb0 <= sMaxWave) {
#pragma unroll
                for (int sub = 0; sub < 2; ++sub) {
                    int tt = pair * 2 + sub;
                    int tb = tb0 + sub * 16;
                    v16bf kf0 = *(const v16bf*)&kb[(tt * 16 + l) * 64 + bOff];
                    v16bf kf1 = *(const v16bf*)&kb[(tt * 16 + l) * 64 + 32 + bOff];
                    v8f c = {};
                    c = WMMA_BF16(qf0, kf0, c);
                    c = WMMA_BF16(qf1, kf1, c);
                    int t = tb + l;
#pragma unroll
                    for (int r = 0; r < 8; ++r) {
                        int s = s0 + half * 8 + r;
                        float w = 0.0f;
                        if (s >= t) w = c[r] * exp2f((float)(s - t) * lg) * invDen[r];
                        wst[wave][(half * 8 + r) * 32 + sub * 16 + l] = (bf16_t)w;
                    }
                }
                // reload 16x32 weight tile as A-fragment (wave-private LDS; DS in-order)
                v16bf wf = make_afrag(&wst[wave][l * 32 + aOff], &wst[wave][l * 32 + aOff + 8]);
#pragma unroll
                for (int dj = 0; dj < 8; ++dj) {
                    v16bf vf = *(const v16bf*)&vb[(dj * 16 + l) * 64 + pair * 32 + bOff];
                    accD[dj] = WMMA_BF16(wf, vf, accD[dj]);
                }
            }
        }
    }

    // write out[b, s, h*HD + d] (f32)
    int b = bh >> 3;
#pragma unroll
    for (int dj = 0; dj < 8; ++dj)
#pragma unroll
        for (int r = 0; r < 8; ++r) {
            int s = s0 + half * 8 + r;
            int d = dj * 16 + l;
            outw[((size_t)b * S_LEN + s) * CF_DIM + h * HD_DIM + d] = accD[dj][r];
        }
}

// ---------------------------------------------------------------------------
// LayerNorm over 1024 channels + SiLU gate, emit bf16 for final GEMM.
// ---------------------------------------------------------------------------
__global__ __launch_bounds__(256)
void ln_gate_kernel(const float* __restrict__ o, const float* __restrict__ g,
                    const float* __restrict__ ln_g, const float* __restrict__ ln_b,
                    bf16_t* __restrict__ yb) {
    const int row = blockIdx.x;
    const int tid = threadIdx.x;
    const int lane = tid & 31, wave = tid >> 5;
    const float* orow = o + (size_t)row * CF_DIM;

    float v[4], s1 = 0.0f, s2 = 0.0f;
#pragma unroll
    for (int k = 0; k < 4; ++k) {
        float x = orow[tid + k * 256];
        v[k] = x; s1 += x; s2 += x * x;
    }
#pragma unroll
    for (int off = 16; off >= 1; off >>= 1) {
        s1 += __shfl_down(s1, off, 32);
        s2 += __shfl_down(s2, off, 32);
    }
    __shared__ float r1[8], r2[8];
    __shared__ float mu_s, rs_s;
    if (lane == 0) { r1[wave] = s1; r2[wave] = s2; }
    __syncthreads();
    if (tid == 0) {
        float t1 = 0.0f, t2 = 0.0f;
#pragma unroll
        for (int w = 0; w < 8; ++w) { t1 += r1[w]; t2 += r2[w]; }
        float mu = t1 * (1.0f / CF_DIM);
        float var = t2 * (1.0f / CF_DIM) - mu * mu;
        mu_s = mu;
        rs_s = rsqrtf(var + 1e-6f);
    }
    __syncthreads();
    float mu = mu_s, rs = rs_s;
#pragma unroll
    for (int k = 0; k < 4; ++k) {
        int i = tid + k * 256;
        float y = (v[k] - mu) * rs * ln_g[i] + ln_b[i];
        float gv = g[(size_t)row * CF_DIM + i];
        float gate = gv / (1.0f + expf(-gv));  // g * sigmoid(g)
        yb[(size_t)row * CF_DIM + i] = (bf16_t)(gate * y);
    }
}

// ---------------------------------------------------------------------------
// Host-side launch
// ---------------------------------------------------------------------------
extern "C" void kernel_launch(void* const* d_in, const int* in_sizes, int n_in,
                              void* d_out, int out_size, void* d_ws, size_t ws_size,
                              hipStream_t stream) {
    const float* x    = (const float*)d_in[0];
    const float* sinp = (const float*)d_in[1];
    const float* cosp = (const float*)d_in[2];
    // d_in[3] = mask  (unused: decay computed analytically)
    const float* Wq = (const float*)d_in[4];
    const float* bq = (const float*)d_in[5];
    const float* Wk = (const float*)d_in[6];
    const float* bk = (const float*)d_in[7];
    const float* Wv = (const float*)d_in[8];
    const float* bv = (const float*)d_in[9];
    const float* Wg = (const float*)d_in[10];
    const float* bg = (const float*)d_in[11];
    const float* Wo = (const float*)d_in[12];
    const float* bo = (const float*)d_in[13];
    const float* lg = (const float*)d_in[14];
    const float* lb = (const float*)d_in[15];

    char* ws = (char*)d_ws;
    // workspace layout (bytes)
    const size_t OFF_XB  = 0;                              // x bf16: 8192*512*2
    const size_t OFF_WQT = OFF_XB  + (size_t)8388608;
    const size_t OFF_WKT = OFF_WQT + (size_t)524288;
    const size_t OFF_WVT = OFF_WKT + (size_t)524288;
    const size_t OFF_WGT = OFF_WVT + (size_t)1048576;
    const size_t OFF_WOT = OFF_WGT + (size_t)1048576;
    const size_t OFF_QR  = OFF_WOT + (size_t)1048576;      // bf16 [b,h,s,kd]
    const size_t OFF_KR  = OFF_QR  + (size_t)8388608;
    const size_t OFF_VRT = OFF_KR  + (size_t)8388608;      // bf16 [b,h,d,s]
    const size_t OFF_G   = OFF_VRT + (size_t)8388608;      // f32  [m,1024]
    const size_t OFF_O   = OFF_G   + (size_t)33554432;     // f32  [m,1024]
    const size_t OFF_YB  = OFF_O   + (size_t)33554432;     // bf16 [m,1024]

    bf16_t* xb  = (bf16_t*)(ws + OFF_XB);
    bf16_t* WqT = (bf16_t*)(ws + OFF_WQT);
    bf16_t* WkT = (bf16_t*)(ws + OFF_WKT);
    bf16_t* WvT = (bf16_t*)(ws + OFF_WVT);
    bf16_t* WgT = (bf16_t*)(ws + OFF_WGT);
    bf16_t* WoT = (bf16_t*)(ws + OFF_WOT);
    bf16_t* qr  = (bf16_t*)(ws + OFF_QR);
    bf16_t* kr  = (bf16_t*)(ws + OFF_KR);
    bf16_t* vrT = (bf16_t*)(ws + OFF_VRT);
    float*  gb  = (float*)(ws + OFF_G);
    float*  ob  = (float*)(ws + OFF_O);
    bf16_t* yb  = (bf16_t*)(ws + OFF_YB);

    // 1) conversions
    {
        int n = M_ROWS * C_DIM;  // 4,194,304
        cvt_f32_bf16<<<(n + 255) / 256, 256, 0, stream>>>(x, xb, n);
        int nw = C_DIM * C_DIM;
        cvt_transpose_bf16<<<(nw + 255) / 256, 256, 0, stream>>>(Wq, WqT, C_DIM, C_DIM);
        cvt_transpose_bf16<<<(nw + 255) / 256, 256, 0, stream>>>(Wk, WkT, C_DIM, C_DIM);
        int nv = C_DIM * CF_DIM;
        cvt_transpose_bf16<<<(nv + 255) / 256, 256, 0, stream>>>(Wv, WvT, C_DIM, CF_DIM);
        cvt_transpose_bf16<<<(nv + 255) / 256, 256, 0, stream>>>(Wg, WgT, C_DIM, CF_DIM);
        cvt_transpose_bf16<<<(nv + 255) / 256, 256, 0, stream>>>(Wo, WoT, CF_DIM, C_DIM);
    }

    // 2) projections
    dim3 blk(256);
    gemm_bf16_wmma<<<dim3(C_DIM / 64, M_ROWS / 128), blk, 0, stream>>>(
        xb, WqT, bq, M_ROWS, C_DIM, C_DIM, 0, sinp, cosp, qr, nullptr);
    gemm_bf16_wmma<<<dim3(C_DIM / 64, M_ROWS / 128), blk, 0, stream>>>(
        xb, WkT, bk, M_ROWS, C_DIM, C_DIM, 1, sinp, cosp, kr, nullptr);
    gemm_bf16_wmma<<<dim3(CF_DIM / 64, M_ROWS / 128), blk, 0, stream>>>(
        xb, WvT, bv, M_ROWS, CF_DIM, C_DIM, 2, nullptr, nullptr, vrT, nullptr);
    gemm_bf16_wmma<<<dim3(CF_DIM / 64, M_ROWS / 128), blk, 0, stream>>>(
        xb, WgT, bg, M_ROWS, CF_DIM, C_DIM, 3, nullptr, nullptr, nullptr, gb);

    // 3) retention attention
    attn_kernel<<<dim3(S_LEN / 64, B_NUM * H_NUM), dim3(128), 0, stream>>>(qr, kr, vrT, ob);

    // 4) layernorm + gate
    ln_gate_kernel<<<dim3(M_ROWS), dim3(256), 0, stream>>>(ob, gb, lg, lb, yb);

    // 5) output projection
    gemm_bf16_wmma<<<dim3(C_DIM / 64, M_ROWS / 128), blk, 0, stream>>>(
        yb, WoT, bo, M_ROWS, C_DIM, CF_DIM, 4, nullptr, nullptr, nullptr, (float*)d_out);

    (void)in_sizes; (void)n_in; (void)out_size; (void)ws_size;
}
